// GRU_with_GCC_28570122453317
// MI455X (gfx1250) — compile-verified, matrix operand order checked by reading, not run on hardware
//
#include <hip/hip_runtime.h>
#include <hip/hip_bf16.h>

// ---------------------------------------------------------------------------
// GRU + gated context conditioning for MI455X (gfx1250, wave32).
//  Phase 0: one-shot f32->bf16 staging of activations/weights into d_ws.
//  Phase 1: LDS-tiled bf16 GEMM (BM=BN=64, BK=32; 4 waves x (2x2) 16x16 tiles,
//           v_wmma_f32_16x16x32_bf16, f32 accumulate). K-tiles are fetched with
//           GLOBAL_LOAD_ASYNC_TO_LDS_B128 into double-buffered LDS so the async
//           engine overlaps the next tile's fetch with the current tile's WMMAs
//           (s_wait_asynccnt 4 = own 4 in-flight chunks of the next tile).
//  Phase 2: 64 sequential step kernels; recurrent GEMM in full-precision
//           v_wmma_f32_16x16x4_f32; h_prev tile staged in LDS; fused gates.
//  gi(48MB)+gf(32MB)+bf16 staging(50MB) in d_ws -> resident in 192MB L2.
// ---------------------------------------------------------------------------

typedef __attribute__((ext_vector_type(16))) __bf16 v16bf;
typedef __attribute__((ext_vector_type(8)))  __bf16 v8bf;
typedef __attribute__((ext_vector_type(8)))  float  v8f;
typedef __attribute__((ext_vector_type(2)))  float  v2f;

#define GB 64       // batch
#define GS 64       // seq len
#define GH 1024     // hidden

#define LDS_STRIDE 40   // bf16 per LDS row (32 data + 8 pad -> bank-safe)

// Async copy: 16B from global (SGPR base + per-lane byte offset) straight into
// LDS (per-lane LDS byte address). Tracked by ASYNCcnt.
__device__ __forceinline__ void async_ld_b128(unsigned lds_off, unsigned goff,
                                              const void* sbase) {
    asm volatile("global_load_async_to_lds_b128 %0, %1, %2"
                 :: "v"(lds_off), "v"(goff), "s"(sbase) : "memory");
}

// ---------------------------------------------------------------------------
// Phase 0: f32 -> bf16 staging (8 elements / thread, b128 in, b128 out)
// ---------------------------------------------------------------------------
__global__ __launch_bounds__(256)
void f32_to_bf16(const float* __restrict__ src, __bf16* __restrict__ dst, int n8) {
    const int i = blockIdx.x * 256 + threadIdx.x;
    if (i < n8) {
        const float4* p = reinterpret_cast<const float4*>(src) + (size_t)i * 2;
        const float4 a = p[0], b = p[1];
        v8bf o;
        o[0]=(__bf16)a.x; o[1]=(__bf16)a.y; o[2]=(__bf16)a.z; o[3]=(__bf16)a.w;
        o[4]=(__bf16)b.x; o[5]=(__bf16)b.y; o[6]=(__bf16)b.z; o[7]=(__bf16)b.w;
        *reinterpret_cast<v8bf*>(dst + (size_t)i * 8) = o;
    }
}

// Fragment from an LDS row: elems 0..7 = row[k0..k0+7], 8..15 = row[k0+16..k0+23]
__device__ __forceinline__ v16bf frag_lds(const __bf16* rowbase, int half) {
    const v8bf lo = *reinterpret_cast<const v8bf*>(rowbase + half * 8);
    const v8bf hi = *reinterpret_cast<const v8bf*>(rowbase + 16 + half * 8);
    return __builtin_shufflevector(lo, hi, 0,1,2,3,4,5,6,7,8,9,10,11,12,13,14,15);
}

// ---------------------------------------------------------------------------
// Phase 1: C[M,N] (+)= A[M,K] * W[N,K]^T + bias[N]    (A, W pre-staged bf16)
// grid: (M/64, N/64); block: 128 threads = 4 waves; wave w -> 32x32 subtile.
// ---------------------------------------------------------------------------
template<bool ACC>
__global__ __launch_bounds__(128)
void gemm_bf16_tiled(const __bf16* __restrict__ A,
                     const __bf16* __restrict__ W,
                     const float*  __restrict__ bias,
                     float* __restrict__ C,
                     int N, int K) {
    __shared__ __bf16 As[2][64 * LDS_STRIDE];   // 2 x 5 KB (double buffer)
    __shared__ __bf16 Bs[2][64 * LDS_STRIDE];   // 2 x 5 KB

    const int lane = threadIdx.x & 31;
    const int wave = threadIdx.x >> 5;
    const int idx  = lane & 15;
    const int half = lane >> 4;
    const int mw   = (wave >> 1) * 32;
    const int nw   = (wave & 1) * 32;

    const __bf16* Abase = A + (size_t)(blockIdx.x * 64) * K;
    const __bf16* Wbase = W + (size_t)(blockIdx.y * 64) * K;

    // Per-thread staging chunks: each thread owns 2 of the 256 16B chunks/matrix.
    const int c1 = threadIdx.x,       r1 = c1 >> 2, q1 = (c1 & 3) * 8;
    const int c2 = threadIdx.x + 128, r2 = c2 >> 2, q2 = (c2 & 3) * 8;
    unsigned lA1[2], lA2[2], lB1[2], lB2[2];
    #pragma unroll
    for (int b = 0; b < 2; ++b) {
        lA1[b] = (unsigned)(uintptr_t)&As[b][r1 * LDS_STRIDE + q1];
        lA2[b] = (unsigned)(uintptr_t)&As[b][r2 * LDS_STRIDE + q2];
        lB1[b] = (unsigned)(uintptr_t)&Bs[b][r1 * LDS_STRIDE + q1];
        lB2[b] = (unsigned)(uintptr_t)&Bs[b][r2 * LDS_STRIDE + q2];
    }

    auto stage = [&](int kt, int buf) {
        const unsigned g1 = (unsigned)((r1 * K + (kt << 5) + q1) * 2);
        const unsigned g2 = (unsigned)((r2 * K + (kt << 5) + q2) * 2);
        async_ld_b128(lA1[buf], g1, Abase);
        async_ld_b128(lA2[buf], g2, Abase);
        async_ld_b128(lB1[buf], g1, Wbase);
        async_ld_b128(lB2[buf], g2, Wbase);
    };

    v8f acc[2][2] = {};

    const int T = K >> 5;               // number of 32-wide K tiles
    stage(0, 0);                        // prologue: tile 0 -> buffer 0
    for (int kt = 0; kt < T; ++kt) {
        const int cur = kt & 1;
        if (kt + 1 < T) {
            stage(kt + 1, cur ^ 1);     // overlap next tile's fetch with compute
            asm volatile("s_wait_asynccnt 0x4" ::: "memory");
        } else {
            asm volatile("s_wait_asynccnt 0x0" ::: "memory");
        }
        __syncthreads();                // all waves' chunks of tile kt landed

        const __bf16* Ab = As[cur];
        const __bf16* Bb = Bs[cur];
        v16bf af[2], bf[2];
        #pragma unroll
        for (int i = 0; i < 2; ++i)
            af[i] = frag_lds(&Ab[(mw + i * 16 + idx) * LDS_STRIDE], half);
        #pragma unroll
        for (int j = 0; j < 2; ++j)
            bf[j] = frag_lds(&Bb[(nw + j * 16 + idx) * LDS_STRIDE], half);

        #pragma unroll
        for (int i = 0; i < 2; ++i)
            #pragma unroll
            for (int j = 0; j < 2; ++j)
                acc[i][j] = __builtin_amdgcn_wmma_f32_16x16x32_bf16(
                                false, af[i], false, bf[j], (short)0, acc[i][j],
                                false, false);
        __syncthreads();                // safe to overwrite buffer cur at kt+2
    }

    // Epilogue: C tile layout -> col = lane%16, row = e + 8*(lane/16)
    #pragma unroll
    for (int i = 0; i < 2; ++i) {
        #pragma unroll
        for (int j = 0; j < 2; ++j) {
            const int ncol = blockIdx.y * 64 + nw + j * 16 + idx;
            const float bv = bias[ncol];
            #pragma unroll
            for (int e = 0; e < 8; ++e) {
                const int mrow = blockIdx.x * 64 + mw + i * 16 + e + 8 * half;
                const size_t off = (size_t)mrow * N + ncol;
                float v = acc[i][j][e] + bv;
                if (ACC) v += C[off];
                C[off] = v;
            }
        }
    }
}

// ---------------------------------------------------------------------------
// Phase 2: one GRU time step. grid (4,16); block 128 = 4 waves.
// Each wave: 16 batch rows x 16 hidden cols; 3 gate GEMMs over K=H, f32 WMMA.
// ---------------------------------------------------------------------------
template<bool FIRST>
__global__ __launch_bounds__(128)
void gru_step(const float* __restrict__ gi,   // [B*S, 3H]
              const float* __restrict__ gf,   // [B*S, 2H]
              const float* __restrict__ Whh,  // [3H, H]
              const float* __restrict__ bhh,  // [3H]
              float* __restrict__ out,        // [B,S,H] + hx tail [B,H]
              int s) {
    __shared__ float ldsH[16 * GH];           // 64 KB: h_prev tile (16 rows x H)

    const int lane = threadIdx.x & 31;
    const int wave = threadIdx.x >> 5;
    const int idx  = lane & 15;
    const int half = lane >> 4;
    const int mt   = blockIdx.x;
    const int nt   = blockIdx.y * 4 + wave;

    if (FIRST) {
        for (int i = threadIdx.x; i < 16 * GH / 4; i += 128)
            reinterpret_cast<float4*>(ldsH)[i] = make_float4(0.f, 0.f, 0.f, 0.f);
    } else {
        for (int i = threadIdx.x; i < 16 * GH / 4; i += 128) {
            const int r = i >> 8;
            const int c = (i & 255) << 2;
            const int b = mt * 16 + r;
            reinterpret_cast<float4*>(ldsH)[i] =
                *reinterpret_cast<const float4*>(out + ((size_t)(b * GS + (s - 1))) * GH + c);
        }
    }
    __syncthreads();

    const int n = nt * 16 + idx;
    const float* wr = Whh + (size_t)(0 * GH + n) * GH;
    const float* wi = Whh + (size_t)(1 * GH + n) * GH;
    const float* wn = Whh + (size_t)(2 * GH + n) * GH;

    v8f ar = {}, ai = {}, an = {};
    #pragma unroll 4
    for (int kk = 0; kk < GH; kk += 4) {
        const int kb = kk + half * 2;
        const float2 av = *reinterpret_cast<const float2*>(&ldsH[idx * GH + kb]);
        const float2 rv = *reinterpret_cast<const float2*>(wr + kb);
        const float2 iv = *reinterpret_cast<const float2*>(wi + kb);
        const float2 nv = *reinterpret_cast<const float2*>(wn + kb);
        v2f a;  a[0] = av.x;  a[1] = av.y;
        v2f br; br[0] = rv.x; br[1] = rv.y;
        v2f bi; bi[0] = iv.x; bi[1] = iv.y;
        v2f bn; bn[0] = nv.x; bn[1] = nv.y;
        ar = __builtin_amdgcn_wmma_f32_16x16x4_f32(false, a, false, br, (short)0, ar, false, false);
        ai = __builtin_amdgcn_wmma_f32_16x16x4_f32(false, a, false, bi, (short)0, ai, false, false);
        an = __builtin_amdgcn_wmma_f32_16x16x4_f32(false, a, false, bn, (short)0, an, false, false);
    }

    const int col = n;
    const float bhr = bhh[col], bhi = bhh[GH + col], bhn = bhh[2 * GH + col];
    #pragma unroll
    for (int e = 0; e < 8; ++e) {
        const int lrow = e + 8 * half;
        const int b    = mt * 16 + lrow;
        const size_t rowI = (size_t)(b * GS + s);
        const float ir  = gi[rowI * (3 * GH) + col];
        const float ii  = gi[rowI * (3 * GH) + GH + col];
        const float inn = gi[rowI * (3 * GH) + 2 * GH + col];
        const float fr  = gf[rowI * (2 * GH) + col];
        const float fi  = gf[rowI * (2 * GH) + GH + col];

        const float hr = ar[e] + bhr;
        const float hi = ai[e] + bhi;
        const float hn = an[e] + bhn;

        const float r  = 1.f / (1.f + __expf(-(ir + hr + fr)));
        const float z  = 1.f / (1.f + __expf(-(ii + hi + fi)));
        const float nn = tanhf(inn + r * hn);
        const float hprev = FIRST ? 0.f : ldsH[lrow * GH + col];
        const float hy = nn + z * (hprev - nn);

        out[rowI * GH + col] = hy;
        if (s == GS - 1)
            out[(size_t)GB * GS * GH + (size_t)b * GH + col] = hy;
    }
}

extern "C" void kernel_launch(void* const* d_in, const int* in_sizes, int n_in,
                              void* d_out, int out_size, void* d_ws, size_t ws_size,
                              hipStream_t stream) {
    (void)in_sizes; (void)n_in; (void)out_size; (void)ws_size;

    const float* input_feats = (const float*)d_in[0];   // [B,S,1024]
    const float* aux0        = (const float*)d_in[1];   // [B,S,1024]
    const float* aux1        = (const float*)d_in[2];   // [B,S,2048]
    const float* W_ih        = (const float*)d_in[3];   // [3H,1024]
    const float* W_hh        = (const float*)d_in[4];   // [3H,H]
    const float* b_ih        = (const float*)d_in[5];   // [3H]
    const float* b_hh        = (const float*)d_in[6];   // [3H]
    const float* W_fh0       = (const float*)d_in[7];   // [2H,1024]
    const float* b_fh0       = (const float*)d_in[8];   // [2H]
    const float* W_fh1       = (const float*)d_in[9];   // [2H,2048]
    const float* b_fh1       = (const float*)d_in[10];  // [2H]
    float* out = (float*)d_out;

    // Workspace layout:
    float*  gi   = (float*)d_ws;                               // [4096,3072] f32
    float*  gf   = gi + (size_t)4096 * 3072;                   // [4096,2048] f32
    __bf16* Xb   = (__bf16*)(gf + (size_t)4096 * 2048);        // [4096,1024]
    __bf16* A0b  = Xb   + (size_t)4096 * 1024;                 // [4096,1024]
    __bf16* A1b  = A0b  + (size_t)4096 * 1024;                 // [4096,2048]
    __bf16* Wihb = A1b  + (size_t)4096 * 2048;                 // [3072,1024]
    __bf16* Wf0b = Wihb + (size_t)3072 * 1024;                 // [2048,1024]
    __bf16* Wf1b = Wf0b + (size_t)2048 * 1024;                 // [2048,2048]

    // Phase 0: stage bf16 copies.
    f32_to_bf16<<<dim3((4096*1024/8)/256), 256, 0, stream>>>(input_feats, Xb,   4096*1024/8);
    f32_to_bf16<<<dim3((4096*1024/8)/256), 256, 0, stream>>>(aux0,        A0b,  4096*1024/8);
    f32_to_bf16<<<dim3((4096*2048/8)/256), 256, 0, stream>>>(aux1,        A1b,  4096*2048/8);
    f32_to_bf16<<<dim3((3072*1024/8)/256), 256, 0, stream>>>(W_ih,        Wihb, 3072*1024/8);
    f32_to_bf16<<<dim3((2048*1024/8)/256), 256, 0, stream>>>(W_fh0,       Wf0b, 2048*1024/8);
    f32_to_bf16<<<dim3((2048*2048/8)/256), 256, 0, stream>>>(W_fh1,       Wf1b, 2048*2048/8);

    // Phase 1: time-independent projections (bf16 WMMA, async double-buffered).
    gemm_bf16_tiled<false><<<dim3(64, 48), 128, 0, stream>>>(Xb,  Wihb, b_ih,  gi, 3072, 1024);
    gemm_bf16_tiled<false><<<dim3(64, 32), 128, 0, stream>>>(A0b, Wf0b, b_fh0, gf, 2048, 1024);
    gemm_bf16_tiled<true ><<<dim3(64, 32), 128, 0, stream>>>(A1b, Wf1b, b_fh1, gf, 2048, 2048);

    // Phase 2: sequential recurrence (full-precision f32 WMMA).
    gru_step<true><<<dim3(4, 16), 128, 0, stream>>>(gi, gf, W_hh, b_hh, out, 0);
    for (int s = 1; s < GS; ++s)
        gru_step<false><<<dim3(4, 16), 128, 0, stream>>>(gi, gf, W_hh, b_hh, out, s);
}
